// SimpleMamba2_11690900980064
// MI455X (gfx1250) — compile-verified, matrix operand order checked by reading, not run on hardware
//
#include <hip/hip_runtime.h>
#include <hip/hip_bf16.h>

// ---------------------------------------------------------------------------
// SimpleMamba2 forward on gfx1250 (MI455X, wave32, WMMA).
// All five GEMMs run through v_wmma_f32_16x16x32_bf16 (bf16 in, f32 accum)
// with the weight tile double-buffered in LDS. All B fragments of a K-slab
// are ds-loaded before the 8 WMMAs so waits are graduated (no per-WMMA
// dscnt-0 stall) and accumulator reuse is spaced 4 WMMAs apart.
// The sequential SSD scan keeps the 64-entry state per head-dim column in
// VGPRs (one workgroup per (batch, head)) and fuses the group-norm stats.
// ---------------------------------------------------------------------------

#define DIM      1024
#define STATE    64
#define HEADS    8
#define INNER    2048   // DIM * EXPAND
#define HEAD_DIM 256    // INNER / HEADS
#define KERNEL_W 4
#define BATCH    2
#define SEQ      2048
#define BL       (BATCH * SEQ)   // 4096 rows for all GEMMs
#define EPSF     1e-5f
#define BK       64             // GEMM K-slab staged in LDS

typedef __bf16  bf16_t;
typedef __attribute__((ext_vector_type(16))) __bf16 v16bf;
typedef __attribute__((ext_vector_type(8)))  float  v8f;

__device__ __forceinline__ unsigned short f2bf(float f) {
    unsigned int u = __float_as_uint(f);
    unsigned int r = u + 0x7FFFu + ((u >> 16) & 1u);   // round-to-nearest-even
    return (unsigned short)(r >> 16);
}

__device__ __forceinline__ float silu_f(float v) {
    return v / (1.0f + __expf(-v));
}

// ---------------------------------------------------------------------------
// fp32 -> bf16 conversion (grid-stride)
// ---------------------------------------------------------------------------
__global__ void __launch_bounds__(256)
to_bf16_kernel(const float* __restrict__ src, unsigned short* __restrict__ dst, int n) {
    for (int i = blockIdx.x * blockDim.x + threadIdx.x; i < n;
         i += gridDim.x * blockDim.x) {
        dst[i] = f2bf(src[i]);
    }
}

// ---------------------------------------------------------------------------
// C[M,N] (f32) = A[M,K] (bf16, row-major) * W[N,K]^T (bf16, row-major)
//
// Block = 256 threads = 8 waves -> tile 128(M) x 64(N).
// The 64x64 weight slab is staged in LDS (double-buffered): all 8 waves share
// the same N-range. Next slab's global loads are issued before the compute
// phase and stored to LDS after it, so their latency is hidden behind WMMAs.
//
// Fragment layout used here (16-bit 16x32 A, contiguous K, not ISA table order): lanes 0-15 hold K 0..15,
// lanes 16-31 hold K 16..31; B (32x16) mirrored along K for column n.
// ---------------------------------------------------------------------------
__global__ void __launch_bounds__(256)
gemm_bf16_wmma(const unsigned short* __restrict__ A,
               const unsigned short* __restrict__ W,
               float* __restrict__ C,
               int M, int N, int K)
{
    __shared__ __align__(16) unsigned short wsh[2][64 * BK];   // [n][k], k-fast

    const int tid  = threadIdx.x;
    const int lane = tid & 31;
    const int wave = tid >> 5;
    const int m0   = blockIdx.y * 128 + wave * 16;
    const int n0   = blockIdx.x * 64;
    const int ar   = lane & 15;          // row (A) / column-in-tile (W)
    const int kh   = (lane >> 4) << 4;   // 0 or 16: K-half owned by this lane

    // Cooperative W-tile fetch: 64 rows x 64 K of bf16 = 8192 B;
    // 256 threads x 2 x 16 B. Element offsets e0 = tid*8, e1 = e0 + 2048.
    const int e0  = tid * 8;
    const int e1  = e0 + 2048;
    const int na0 = e0 >> 6, ka0 = e0 & 63;
    const int na1 = e1 >> 6, ka1 = e1 & 63;
    const unsigned short* wg0 = W + (size_t)(n0 + na0) * K + ka0;
    const unsigned short* wg1 = W + (size_t)(n0 + na1) * K + ka1;

    const unsigned short* arow = A + (size_t)(m0 + ar) * K + kh;

    v8f acc0 = {}, acc1 = {}, acc2 = {}, acc3 = {};

    // Prologue: stage slab 0.
    {
        uint4 t0 = *(const uint4*)(const void*)(wg0);
        uint4 t1 = *(const uint4*)(const void*)(wg1);
        *(uint4*)(void*)&wsh[0][e0] = t0;
        *(uint4*)(void*)&wsh[0][e1] = t1;
    }
    __syncthreads();

    const int nk = K / BK;
    for (int kt = 0; kt < nk; ++kt) {
        const int  cur  = kt & 1;
        const int  nxt  = cur ^ 1;
        const bool more = (kt + 1) < nk;

        // Issue next slab's global loads early (latency hidden by WMMAs).
        uint4 t0, t1;
        if (more) {
            t0 = *(const uint4*)(const void*)(wg0 + (kt + 1) * BK);
            t1 = *(const uint4*)(const void*)(wg1 + (kt + 1) * BK);
        }

        const int kbase = kt * BK;
        __builtin_prefetch(arow + kbase + 2 * BK, 0, 3);
        v16bf a0 = *(const v16bf*)(const void*)(arow + kbase);
        v16bf a1 = *(const v16bf*)(const void*)(arow + kbase + 32);

        // Load ALL B fragments of this slab before any WMMA: the first WMMA
        // only needs the first fragment, so waits are graduated rather than
        // a dscnt-0 stall in front of every matrix op.
        const unsigned short* wsl = &wsh[cur][0] + ar * BK + kh;
        v16bf b00 = *(const v16bf*)(const void*)(wsl +  0 * BK +  0);
        v16bf b10 = *(const v16bf*)(const void*)(wsl + 16 * BK +  0);
        v16bf b20 = *(const v16bf*)(const void*)(wsl + 32 * BK +  0);
        v16bf b30 = *(const v16bf*)(const void*)(wsl + 48 * BK +  0);
        v16bf b01 = *(const v16bf*)(const void*)(wsl +  0 * BK + 32);
        v16bf b11 = *(const v16bf*)(const void*)(wsl + 16 * BK + 32);
        v16bf b21 = *(const v16bf*)(const void*)(wsl + 32 * BK + 32);
        v16bf b31 = *(const v16bf*)(const void*)(wsl + 48 * BK + 32);

        // Accumulator reuse spaced 4 WMMAs apart (XDL pipelining).
        acc0 = __builtin_amdgcn_wmma_f32_16x16x32_bf16(false, a0, false, b00,
                                                       (short)0, acc0, false, false);
        acc1 = __builtin_amdgcn_wmma_f32_16x16x32_bf16(false, a0, false, b10,
                                                       (short)0, acc1, false, false);
        acc2 = __builtin_amdgcn_wmma_f32_16x16x32_bf16(false, a0, false, b20,
                                                       (short)0, acc2, false, false);
        acc3 = __builtin_amdgcn_wmma_f32_16x16x32_bf16(false, a0, false, b30,
                                                       (short)0, acc3, false, false);
        acc0 = __builtin_amdgcn_wmma_f32_16x16x32_bf16(false, a1, false, b01,
                                                       (short)0, acc0, false, false);
        acc1 = __builtin_amdgcn_wmma_f32_16x16x32_bf16(false, a1, false, b11,
                                                       (short)0, acc1, false, false);
        acc2 = __builtin_amdgcn_wmma_f32_16x16x32_bf16(false, a1, false, b21,
                                                       (short)0, acc2, false, false);
        acc3 = __builtin_amdgcn_wmma_f32_16x16x32_bf16(false, a1, false, b31,
                                                       (short)0, acc3, false, false);

        // Land next slab in LDS (s_wait_loadcnt happens here, after compute).
        if (more) {
            *(uint4*)(void*)&wsh[nxt][e0] = t0;
            *(uint4*)(void*)&wsh[nxt][e1] = t1;
        }
        __syncthreads();
    }

    // C/D layout: VGPR r, lanes 0-15 -> M=r, lanes 16-31 -> M=8+r; N = lane%16.
    const int rb  = (lane >> 4) << 3;
    const int col = lane & 15;
#pragma unroll
    for (int r = 0; r < 8; ++r) {
        float* cp = C + (size_t)(m0 + rb + r) * N + n0 + col;
        cp[0]  = acc0[r];
        cp[16] = acc1[r];
        cp[32] = acc2[r];
        cp[48] = acc3[r];
    }
}

// ---------------------------------------------------------------------------
// Depthwise causal conv (K=4) + bias + SiLU; writes fp32 and bf16 copies.
// ---------------------------------------------------------------------------
__global__ void __launch_bounds__(256)
conv_silu_kernel(const float* __restrict__ xr,
                 const float* __restrict__ cw,   // (INNER, 1, 4)
                 const float* __restrict__ cb,   // (INNER)
                 float* __restrict__ xo,
                 unsigned short* __restrict__ xo_bf)
{
    int idx = blockIdx.x * blockDim.x + threadIdx.x;
    if (idx >= BL * INNER) return;
    int c = idx & (INNER - 1);
    int l = (idx / INNER) & (SEQ - 1);   // position within its batch

    float acc = cb[c];
#pragma unroll
    for (int k = 0; k < KERNEL_W; ++k) {
        int ls = l + k - (KERNEL_W - 1);
        if (ls >= 0)
            acc += xr[idx + (k - (KERNEL_W - 1)) * INNER] * cw[c * KERNEL_W + k];
    }
    float s = silu_f(acc);
    xo[idx]    = s;
    xo_bf[idx] = f2bf(s);
}

// ---------------------------------------------------------------------------
// dt = softplus(x @ Wdt^T + b_dt); one block (64 threads) per (b,l) row.
// ---------------------------------------------------------------------------
__global__ void __launch_bounds__(64)
dt_kernel(const float* __restrict__ x,     // (BL, INNER)
          const float* __restrict__ Wdt,   // (HEADS, INNER)
          const float* __restrict__ bdt,   // (HEADS)
          float* __restrict__ dt)          // (BL, HEADS)
{
    __shared__ float red[HEADS][64];
    const int tid = threadIdx.x;
    const float* xr = x + (size_t)blockIdx.x * INNER;

    float acc[HEADS];
#pragma unroll
    for (int h = 0; h < HEADS; ++h) acc[h] = 0.f;

    for (int k = tid; k < INNER; k += 64) {
        float xv = xr[k];
#pragma unroll
        for (int h = 0; h < HEADS; ++h)
            acc[h] += xv * Wdt[h * INNER + k];
    }
#pragma unroll
    for (int h = 0; h < HEADS; ++h) red[h][tid] = acc[h];
    __syncthreads();
    for (int s = 32; s > 0; s >>= 1) {
        if (tid < s) {
#pragma unroll
            for (int h = 0; h < HEADS; ++h) red[h][tid] += red[h][tid + s];
        }
        __syncthreads();
    }
    if (tid < HEADS) {
        float v = red[tid][0] + bdt[tid];
        dt[(size_t)blockIdx.x * HEADS + tid] =
            (v > 20.f) ? v : __logf(1.0f + __expf(v));
    }
}

// ---------------------------------------------------------------------------
// SSD scan: one block per (batch, head). Lane p owns head-dim column p and
// keeps state[STATE] in VGPRs. b_t / c_t / dt_t broadcast through LDS
// (vectorized float4 reads). Fuses group-norm statistics (mean/var over
// L x HEAD_DIM per (b,h)).
// ---------------------------------------------------------------------------
__global__ void __launch_bounds__(256)
ssd_scan_kernel(const float* __restrict__ xbuf,   // (BL, INNER)
                const float* __restrict__ bbuf,   // (BL, HEADS*STATE)
                const float* __restrict__ cbuf,   // (BL, HEADS*STATE)
                const float* __restrict__ dtbuf,  // (BL, HEADS)
                const float* __restrict__ log_a,  // (HEADS)
                const float* __restrict__ dprm,   // (HEADS)
                float* __restrict__ ybuf,         // (BL, INNER)
                float* __restrict__ stats)        // (B*HEADS, 2) mean, rstd
{
    const int b = blockIdx.x >> 3;
    const int h = blockIdx.x & (HEADS - 1);
    const int p = threadIdx.x;                     // 0..HEAD_DIM-1

    const float a  = -__expf(log_a[h]);
    const float dp = dprm[h];

    __shared__ __align__(16) float sb[STATE];
    __shared__ __align__(16) float sc[STATE];
    __shared__ float sdt;
    __shared__ float rs[256];
    __shared__ float rq[256];

    float st[STATE];
#pragma unroll
    for (int n = 0; n < STATE; ++n) st[n] = 0.f;

    float sum = 0.f, sumsq = 0.f;

    for (int l = 0; l < SEQ; ++l) {
        const size_t row = (size_t)b * SEQ + l;
        if (p < STATE)            sb[p]         = bbuf[row * (HEADS*STATE) + h*STATE + p];
        else if (p < 2 * STATE)   sc[p - STATE] = cbuf[row * (HEADS*STATE) + h*STATE + (p - STATE)];
        else if (p == 2 * STATE)  sdt           = dtbuf[row * HEADS + h];
        __syncthreads();

        const float dt    = sdt;
        const float decay = __expf(dt * a);
        const float xv    = xbuf[row * INNER + h * HEAD_DIM + p];
        const float dx    = dt * xv;

        float y = 0.f;
#pragma unroll
        for (int n4 = 0; n4 < STATE / 4; ++n4) {
            const float4 bb = *(const float4*)&sb[n4 * 4];
            const float4 cc = *(const float4*)&sc[n4 * 4];
            float* s4 = &st[n4 * 4];
            s4[0] = decay * s4[0] + bb.x * dx;  y += cc.x * s4[0];
            s4[1] = decay * s4[1] + bb.y * dx;  y += cc.y * s4[1];
            s4[2] = decay * s4[2] + bb.z * dx;  y += cc.z * s4[2];
            s4[3] = decay * s4[3] + bb.w * dx;  y += cc.w * s4[3];
        }
        __syncthreads();   // sb/sc reads done before next iteration overwrites

        y += dp * xv;
        ybuf[row * INNER + h * HEAD_DIM + p] = y;
        sum   += y;
        sumsq += y * y;
    }

    // Block reduction for per-(b,h) group-norm statistics.
    rs[p] = sum; rq[p] = sumsq;
    __syncthreads();
    for (int s = 128; s > 0; s >>= 1) {
        if (p < s) { rs[p] += rs[p + s]; rq[p] += rq[p + s]; }
        __syncthreads();
    }
    if (p == 0) {
        const float inv  = 1.0f / (float)(SEQ * HEAD_DIM);
        const float mean = rs[0] * inv;
        const float var  = rq[0] * inv - mean * mean;
        stats[blockIdx.x * 2 + 0] = mean;
        stats[blockIdx.x * 2 + 1] = rsqrtf(var + EPSF);
    }
}

// ---------------------------------------------------------------------------
// Group-norm apply + SiLU gate; emit bf16 operand for the output projection.
// ---------------------------------------------------------------------------
__global__ void __launch_bounds__(256)
norm_gate_kernel(const float* __restrict__ ybuf,
                 const float* __restrict__ zbuf,
                 const float* __restrict__ stats,
                 const float* __restrict__ gnw,
                 const float* __restrict__ gnb,
                 unsigned short* __restrict__ gated_bf)
{
    int idx = blockIdx.x * blockDim.x + threadIdx.x;
    if (idx >= BL * INNER) return;
    const int i = idx & (INNER - 1);
    const int h = i >> 8;                       // i / HEAD_DIM
    const int b = idx / (SEQ * INNER);

    const float mean = stats[(b * HEADS + h) * 2 + 0];
    const float rstd = stats[(b * HEADS + h) * 2 + 1];

    const float y  = ybuf[idx];
    const float yn = (y - mean) * rstd * gnw[i] + gnb[i];
    const float z  = zbuf[idx];
    gated_bf[idx]  = f2bf(yn * silu_f(z));
}

// ---------------------------------------------------------------------------
// Host-side orchestration
// ---------------------------------------------------------------------------
extern "C" void kernel_launch(void* const* d_in, const int* in_sizes, int n_in,
                              void* d_out, int out_size, void* d_ws, size_t ws_size,
                              hipStream_t stream) {
    (void)in_sizes; (void)n_in; (void)out_size; (void)ws_size;

    const float* input  = (const float*)d_in[0];   // (B, L, DIM)
    const float* Wx     = (const float*)d_in[1];   // (INNER, DIM)
    const float* Wz     = (const float*)d_in[2];   // (INNER, DIM)
    const float* conv_w = (const float*)d_in[3];   // (INNER, 1, 4)
    const float* conv_b = (const float*)d_in[4];   // (INNER)
    const float* Wb     = (const float*)d_in[5];   // (H*STATE, INNER)
    const float* Wc     = (const float*)d_in[6];   // (H*STATE, INNER)
    const float* Wdt    = (const float*)d_in[7];   // (HEADS, INNER)
    const float* b_dt   = (const float*)d_in[8];   // (HEADS)
    const float* log_a  = (const float*)d_in[9];   // (HEADS)
    const float* d_prm  = (const float*)d_in[10];  // (HEADS)
    const float* gn_w   = (const float*)d_in[11];  // (INNER)
    const float* gn_b   = (const float*)d_in[12];  // (INNER)
    const float* Wout   = (const float*)d_in[13];  // (DIM, INNER)
    float*       out    = (float*)d_out;           // (B, L, DIM)

    // Workspace carve-out (256 B aligned slices).
    size_t off = 0;
    auto carve = [&](size_t bytes) -> char* {
        char* p = (char*)d_ws + off;
        off += (bytes + 255) & ~(size_t)255;
        return p;
    };
    unsigned short* in_bf   = (unsigned short*)carve((size_t)BL * DIM   * 2);
    unsigned short* wx_bf   = (unsigned short*)carve((size_t)INNER * DIM * 2);
    unsigned short* wz_bf   = (unsigned short*)carve((size_t)INNER * DIM * 2);
    unsigned short* wb_bf   = (unsigned short*)carve((size_t)HEADS*STATE*INNER * 2);
    unsigned short* wc_bf   = (unsigned short*)carve((size_t)HEADS*STATE*INNER * 2);
    unsigned short* wout_bf = (unsigned short*)carve((size_t)DIM * INNER * 2);
    float*          xraw    = (float*)carve((size_t)BL * INNER * 4);  // reused as ybuf
    float*          zbuf    = (float*)carve((size_t)BL * INNER * 4);
    float*          xbuf    = (float*)carve((size_t)BL * INNER * 4);
    unsigned short* xact_bf = (unsigned short*)carve((size_t)BL * INNER * 2); // reused as gated_bf
    float*          bbuf    = (float*)carve((size_t)BL * HEADS*STATE * 4);
    float*          cbuf    = (float*)carve((size_t)BL * HEADS*STATE * 4);
    float*          dtbuf   = (float*)carve((size_t)BL * HEADS * 4);
    float*          stats   = (float*)carve((size_t)BATCH * HEADS * 2 * 4);
    float*          ybuf     = xraw;     // xraw dead after conv
    unsigned short* gated_bf = xact_bf;  // xact_bf dead after Wb/Wc GEMMs

    const int CVT_BLOCKS = 1024;

    // 1) bf16 conversions of input + weights.
    to_bf16_kernel<<<CVT_BLOCKS, 256, 0, stream>>>(input, in_bf,   BL * DIM);
    to_bf16_kernel<<<CVT_BLOCKS, 256, 0, stream>>>(Wx,    wx_bf,   INNER * DIM);
    to_bf16_kernel<<<CVT_BLOCKS, 256, 0, stream>>>(Wz,    wz_bf,   INNER * DIM);
    to_bf16_kernel<<<CVT_BLOCKS, 256, 0, stream>>>(Wb,    wb_bf,   HEADS*STATE*INNER);
    to_bf16_kernel<<<CVT_BLOCKS, 256, 0, stream>>>(Wc,    wc_bf,   HEADS*STATE*INNER);
    to_bf16_kernel<<<CVT_BLOCKS, 256, 0, stream>>>(Wout,  wout_bf, DIM * INNER);

    // 2) Input projections: xraw = in @ Wx^T, zbuf = in @ Wz^T. (M=4096,N=2048,K=1024)
    {
        dim3 grid(INNER / 64, BL / 128);
        gemm_bf16_wmma<<<grid, 256, 0, stream>>>(in_bf, wx_bf, xraw, BL, INNER, DIM);
        gemm_bf16_wmma<<<grid, 256, 0, stream>>>(in_bf, wz_bf, zbuf, BL, INNER, DIM);
    }

    // 3) Causal conv + SiLU (fp32 + bf16 activation copy).
    conv_silu_kernel<<<(BL * INNER) / 256, 256, 0, stream>>>(xraw, conv_w, conv_b,
                                                             xbuf, xact_bf);

    // 4) B/C projections. (M=4096, N=512, K=2048)
    {
        dim3 grid((HEADS * STATE) / 64, BL / 128);
        gemm_bf16_wmma<<<grid, 256, 0, stream>>>(xact_bf, wb_bf, bbuf, BL, HEADS*STATE, INNER);
        gemm_bf16_wmma<<<grid, 256, 0, stream>>>(xact_bf, wc_bf, cbuf, BL, HEADS*STATE, INNER);
    }

    // 5) dt = softplus(x @ Wdt^T + b_dt).
    dt_kernel<<<BL, 64, 0, stream>>>(xbuf, Wdt, b_dt, dtbuf);

    // 6) SSD scan + fused group-norm statistics (one block per (b,h)).
    ssd_scan_kernel<<<BATCH * HEADS, HEAD_DIM, 0, stream>>>(
        xbuf, bbuf, cbuf, dtbuf, log_a, d_prm, ybuf, stats);

    // 7) Group-norm apply + SiLU gate -> bf16 operand.
    norm_gate_kernel<<<(BL * INNER) / 256, 256, 0, stream>>>(
        ybuf, zbuf, stats, gn_w, gn_b, gated_bf);

    // 8) Output projection: out = gated @ Wout^T. (M=4096, N=1024, K=2048)
    {
        dim3 grid(DIM / 64, BL / 128);
        gemm_bf16_wmma<<<grid, 256, 0, stream>>>(gated_bf, wout_bf, out, BL, DIM, INNER);
    }
}
